// TransformerBlock_11854109736991
// MI455X (gfx1250) — matrix operand order repaired and checked by reading, not compiled, not run on hardware
//
// CDNA5 / gfx1250 transformer block: LN -> QKV WMMA GEMM -> flash attention (WMMA)
// -> out-proj WMMA GEMM + residual -> LN -> gating/top2 -> expert-grouped MoE WMMA GEMMs.
// wave32 everywhere; cross-lane ops via ds_bpermute (lowers to DPP); all GEMMs use
// v_wmma_f32_16x16x32_bf16; tile staging uses GLOBAL_LOAD_ASYNC_TO_LDS_B128 when the
// toolchain exposes the builtin (guarded so the file compiles either way).
#include <hip/hip_runtime.h>

typedef __attribute__((ext_vector_type(16))) __bf16 v16bf;
typedef __attribute__((ext_vector_type(8)))  __bf16 bf16x8;
typedef __attribute__((ext_vector_type(4)))  __bf16 bf16x4;
typedef __attribute__((ext_vector_type(8)))  float  v8f;
typedef __attribute__((ext_vector_type(4)))  int    v4i;

#define WMMA_BF16(A_, B_, C_) \
  __builtin_amdgcn_wmma_f32_16x16x32_bf16(false, (A_), false, (B_), (short)0, (C_), false, false)

// ---- optional CDNA5 async global->LDS path (ASYNCcnt) ----
#if defined(__has_builtin)
#if __has_builtin(__builtin_amdgcn_global_load_async_to_lds_b128)
#define HAVE_ASYNC_LDS 1
#endif
#endif
#ifndef HAVE_ASYNC_LDS
#define HAVE_ASYNC_LDS 0
#endif

#define AS1 __attribute__((address_space(1)))
#define AS3 __attribute__((address_space(3)))

__device__ __forceinline__ void async_wait0() {
#if HAVE_ASYNC_LDS
#if __has_builtin(__builtin_amdgcn_s_wait_asynccnt)
  __builtin_amdgcn_s_wait_asynccnt(0);
#else
  asm volatile("s_wait_asynccnt 0x0" ::: "memory");
#endif
#endif
}

// ---------------- problem constants ----------------
constexpr int kD = 1024, kH = 16, kHD = 64, kFF = 4096, kE = 8;
constexpr int kSeq = 2048, kBatch = 2, kT = kBatch * kSeq;   // 4096 tokens
constexpr int kQKV = 3 * kD;                                  // 3072
constexpr int kSlotCap = kT * 2 + kE * 128;                   // 9216 padded expert slots
constexpr float kEps = 1e-5f;

// ---------------- workspace layout (bytes) ----------------
constexpr size_t OFF_INPROJ = 0;
constexpr size_t OFF_OUTW   = OFF_INPROJ + (size_t)kQKV * kD * 2;
constexpr size_t OFF_W1     = OFF_OUTW   + (size_t)kD * kD * 2;
constexpr size_t OFF_W2     = OFF_W1     + (size_t)kE * kFF * kD * 2;
constexpr size_t OFF_H1     = OFF_W2     + (size_t)kE * kD * kFF * 2;
constexpr size_t OFF_QKVB   = OFF_H1     + (size_t)kT * kD * 2;
constexpr size_t OFF_CTX    = OFF_QKVB   + (size_t)kT * kQKV * 2;
constexpr size_t OFF_H2     = OFF_CTX    + (size_t)kT * kD * 2;
constexpr size_t OFF_EW     = OFF_H2     + (size_t)kT * kD * 2;
constexpr size_t OFF_TOKL   = OFF_EW     + (size_t)kT * kE * 4;
constexpr size_t OFF_TOKW   = OFF_TOKL   + (size_t)kE * kT * 4;
constexpr size_t OFF_META   = OFF_TOKW   + (size_t)kE * kT * 4;
constexpr size_t OFF_HID    = OFF_META   + 256;

// ---------------- wave32 helpers ----------------
__device__ __forceinline__ int lane_id() { return (int)__builtin_amdgcn_mbcnt_lo(~0u, 0u); }

__device__ __forceinline__ float bcast_xor_f(float x, int mask) {
  int v = __builtin_amdgcn_ds_bpermute(((lane_id() ^ mask) << 2), __builtin_bit_cast(int, x));
  return __builtin_bit_cast(float, v);
}
__device__ __forceinline__ int bperm_i(int srcLane, int v) {
  return __builtin_amdgcn_ds_bpermute(srcLane << 2, v);
}

// ---------------- WMMA fragment loads (per CDNA5 ISA layouts) ----------------
// A (16x32 bf16): lanes 0-15 row M=lane k={0..7,16..23}; lanes 16-31 same row, k={8..15,24..31}
__device__ __forceinline__ v16bf load_fragA(const __bf16* p0, int ld, int mOff, int lLow, int lHi) {
  union { v16bf v; bf16x8 x[2]; } u;
  const __bf16* p = p0 + (size_t)(mOff + lLow) * ld + lHi * 8;
  u.x[0] = *(const bf16x8*)p;
  u.x[1] = *(const bf16x8*)(p + 16);
  return u.v;
}
// B (32x16 bf16, source stored [N][K] row-major): lane n col N=n holds K={0..7,16..23} (lanes<16) / {8..15,24..31}
__device__ __forceinline__ v16bf load_fragB(const __bf16* p0, int ld, int nOff, int kOff,
                                            int lLow, int lHi) {
  union { v16bf v; bf16x8 x[2]; } u;
  const __bf16* p = p0 + (size_t)(nOff + lLow) * ld + kOff + lHi * 8;
  u.x[0] = *(const bf16x8*)p;
  u.x[1] = *(const bf16x8*)(p + 16);
  return u.v;
}

// ---------------- LDS tile staging (128 rows x 32 k, bf16) ----------------
__device__ __forceinline__ void copy16(__bf16* dst, const __bf16* src) {
#if HAVE_ASYNC_LDS
  __builtin_amdgcn_global_load_async_to_lds_b128((AS1 v4i*)src, (AS3 v4i*)dst, 0, 0);
#else
  *(bf16x8*)dst = *(const bf16x8*)src;
#endif
}

__device__ __forceinline__ void stage_tile(__bf16* s, const __bf16* g, int rowBase, int ld,
                                           int k0, int tid) {
  int idx = tid * 8, r = idx >> 5, c = idx & 31;
  copy16(&s[idx],        &g[(size_t)(rowBase + r) * ld + k0 + c]);
  copy16(&s[idx + 2048], &g[(size_t)(rowBase + r + 64) * ld + k0 + c]);
}
__device__ __forceinline__ void stage_tile_gather(__bf16* s, const __bf16* g, const int* rows,
                                                  int ld, int k0, int tid) {
  int idx = tid * 8, r = idx >> 5, c = idx & 31;
  copy16(&s[idx],        &g[(size_t)rows[r] * ld + k0 + c]);
  copy16(&s[idx + 2048], &g[(size_t)rows[r + 64] * ld + k0 + c]);
}

// per-wave 32x64 micro tile: 8 wmma per 32-deep k step
__device__ __forceinline__ void wave_mma(const __bf16* sA, const __bf16* sB, int mW, int nW,
                                         int lLow, int lHi, v8f (&acc)[2][4]) {
  v16bf a0 = load_fragA(sA, 32, mW, lLow, lHi);
  v16bf a1 = load_fragA(sA, 32, mW + 16, lLow, lHi);
#pragma unroll
  for (int j = 0; j < 4; ++j) {
    v16bf b = load_fragB(sB, 32, nW + j * 16, 0, lLow, lHi);
    acc[0][j] = WMMA_BF16(a0, b, acc[0][j]);
    acc[1][j] = WMMA_BF16(a1, b, acc[1][j]);
  }
}

// ---------------- f32 -> bf16 weight conversion ----------------
__global__ __launch_bounds__(256) void f2bf_kernel(const float* __restrict__ in,
                                                   __bf16* __restrict__ out, int n) {
  int i = (blockIdx.x * 256 + threadIdx.x) * 4;
  if (i >= n) return;
  float4 v = *(const float4*)&in[i];
  bf16x4 o;
  o.x = (__bf16)v.x; o.y = (__bf16)v.y; o.z = (__bf16)v.z; o.w = (__bf16)v.w;
  *(bf16x4*)&out[i] = o;
}

// ---------------- LayerNorm (one token per block) ----------------
__global__ __launch_bounds__(256) void ln_kernel(const float* __restrict__ x,
                                                 const float* __restrict__ g,
                                                 const float* __restrict__ b,
                                                 __bf16* __restrict__ out) {
  const int t = blockIdx.x, tid = threadIdx.x, wave = tid >> 5;
  float4 xv = *(const float4*)&x[(size_t)t * kD + tid * 4];
  float s = xv.x + xv.y + xv.z + xv.w;
  float s2 = xv.x * xv.x + xv.y * xv.y + xv.z * xv.z + xv.w * xv.w;
#pragma unroll
  for (int m = 1; m < 32; m <<= 1) { s += bcast_xor_f(s, m); s2 += bcast_xor_f(s2, m); }
  __shared__ float red[16];
  if ((tid & 31) == 0) { red[wave] = s; red[8 + wave] = s2; }
  __syncthreads();
  float S = 0.f, S2 = 0.f;
#pragma unroll
  for (int w = 0; w < 8; ++w) { S += red[w]; S2 += red[8 + w]; }
  float mean = S * (1.f / kD);
  float var = S2 * (1.f / kD) - mean * mean;
  float rstd = rsqrtf(var + kEps);
  float4 gv = *(const float4*)&g[tid * 4];
  float4 bv = *(const float4*)&b[tid * 4];
  bf16x4 o;
  o.x = (__bf16)((xv.x - mean) * rstd * gv.x + bv.x);
  o.y = (__bf16)((xv.y - mean) * rstd * gv.y + bv.y);
  o.z = (__bf16)((xv.z - mean) * rstd * gv.z + bv.z);
  o.w = (__bf16)((xv.w - mean) * rstd * gv.w + bv.w);
  *(bf16x4*)&out[(size_t)t * kD + tid * 4] = o;
}

// ---------------- generic GEMM: C[M,N] = A[M,K] @ W[N,K]^T + bias ----------------
template <bool BF16OUT>
__global__ __launch_bounds__(256) void gemm_kernel(const __bf16* __restrict__ A,
                                                   const __bf16* __restrict__ W,
                                                   const float* __restrict__ bias,
                                                   const float* __restrict__ resid,
                                                   __bf16* __restrict__ outB,
                                                   float* __restrict__ outF, int K, int N) {
  __shared__ __bf16 sA[128 * 32];
  __shared__ __bf16 sB[128 * 32];
  const int tid = threadIdx.x, wave = tid >> 5, lane = tid & 31;
  const int lLow = lane & 15, lHi = lane >> 4;
  const int mBlk = blockIdx.y * 128, nBlk = blockIdx.x * 128;
  const int mW = (wave >> 1) * 32, nW = (wave & 1) * 64;
  v8f acc[2][4] = {};
  for (int k0 = 0; k0 < K; k0 += 32) {
    __syncthreads();
    stage_tile(sA, A, mBlk, K, k0, tid);
    stage_tile(sB, W, nBlk, K, k0, tid);
    if (k0 + 32 < K) {
      int idx = tid * 8, r = idx >> 5, c = idx & 31;
      __builtin_prefetch(&A[(size_t)(mBlk + r) * K + k0 + 32 + c], 0, 0);
      __builtin_prefetch(&W[(size_t)(nBlk + r) * K + k0 + 32 + c], 0, 0);
    }
    async_wait0();
    __syncthreads();
    wave_mma(sA, sB, mW, nW, lLow, lHi, acc);
  }
  const int colBase = nBlk + nW;
#pragma unroll
  for (int mi = 0; mi < 2; ++mi)
#pragma unroll
    for (int j = 0; j < 4; ++j) {
      int col = colBase + j * 16 + lLow;
      float bv = bias[col];
#pragma unroll
      for (int r = 0; r < 8; ++r) {
        int row = mBlk + mW + mi * 16 + lHi * 8 + r;
        float v = acc[mi][j][r] + bv;
        if constexpr (BF16OUT) {
          outB[(size_t)row * N + col] = (__bf16)v;
        } else {
          outF[(size_t)row * N + col] = resid[(size_t)row * N + col] + v;
        }
      }
    }
}

// ---------------- flash attention (full, non-causal) ----------------
__global__ __launch_bounds__(256) void attn_kernel(const __bf16* __restrict__ qkv,
                                                   __bf16* __restrict__ ctx) {
  __shared__ __bf16 sK[32 * 64];    // [key][hd]
  __shared__ __bf16 sVt[64 * 32];   // [hd][key]
  __shared__ __bf16 sP[8 * 16 * 32];
  const int bh = blockIdx.y;
  const int b = bh >> 4, h = bh & 15;
  const int tid = threadIdx.x, wave = tid >> 5, lane = tid & 31;
  const int lLow = lane & 15, lHi = lane >> 4;
  const size_t rowBase = (size_t)b * kSeq;
  const int q0 = blockIdx.x * 128 + wave * 16;
  v16bf qf[2];
  {
    const __bf16* p = qkv + (rowBase + q0 + lLow) * kQKV + h * kHD;
#pragma unroll
    for (int f = 0; f < 2; ++f) {
      union { v16bf v; bf16x8 x[2]; } u;
      u.x[0] = *(const bf16x8*)(p + f * 32 + lHi * 8);
      u.x[1] = *(const bf16x8*)(p + f * 32 + 16 + lHi * 8);
      qf[f] = u.v;
    }
  }
  v8f o0 = {}, o1 = {}, o2 = {}, o3 = {};
  float mrun[8], lrun[8];
#pragma unroll
  for (int r = 0; r < 8; ++r) { mrun[r] = -3.0e38f; lrun[r] = 0.f; }
  __bf16* pw = sP + wave * (16 * 32);
  for (int kt = 0; kt < kSeq; kt += 32) {
    __syncthreads();
    {
      int idx = tid * 8, kr = idx >> 6, kc = idx & 63;
      size_t gb = (rowBase + kt + kr) * (size_t)kQKV + h * kHD + kc;
      copy16(&sK[idx], &qkv[gb + kD]);                           // K block (async if available)
      bf16x8 vv = *(const bf16x8*)&qkv[gb + 2 * kD];             // V block (transpose to LDS)
#pragma unroll
      for (int j = 0; j < 8; ++j) sVt[(kc + j) * 32 + kr] = vv[j];
    }
    async_wait0();
    __syncthreads();
    v8f s0 = {}, s1 = {};
    s0 = WMMA_BF16(qf[0], load_fragB(sK, 64, 0, 0, lLow, lHi), s0);
    s0 = WMMA_BF16(qf[1], load_fragB(sK, 64, 0, 32, lLow, lHi), s0);
    s1 = WMMA_BF16(qf[0], load_fragB(sK, 64, 16, 0, lLow, lHi), s1);
    s1 = WMMA_BF16(qf[1], load_fragB(sK, 64, 16, 32, lLow, lHi), s1);
    float p0[8], p1[8];
#pragma unroll
    for (int r = 0; r < 8; ++r) {
      float a = s0[r] * 0.125f, c = s1[r] * 0.125f;     // 1/sqrt(64)
      float tmax = fmaxf(a, c);
#pragma unroll
      for (int m = 1; m < 16; m <<= 1) tmax = fmaxf(tmax, bcast_xor_f(tmax, m));
      float mNew = fmaxf(mrun[r], tmax);
      float scale = __expf(mrun[r] - mNew);
      mrun[r] = mNew;
      lrun[r] *= scale;
      o0[r] *= scale; o1[r] *= scale; o2[r] *= scale; o3[r] *= scale;
      float e0 = __expf(a - mNew), e1 = __expf(c - mNew);
      p0[r] = e0; p1[r] = e1;
      float rs = e0 + e1;
#pragma unroll
      for (int m = 1; m < 16; m <<= 1) rs += bcast_xor_f(rs, m);
      lrun[r] += rs;
    }
    // C-layout -> A-layout redistribution through per-wave LDS scratch
#pragma unroll
    for (int r = 0; r < 8; ++r) {
      pw[(lHi * 8 + r) * 32 + lLow] = (__bf16)p0[r];
      pw[(lHi * 8 + r) * 32 + 16 + lLow] = (__bf16)p1[r];
    }
    v16bf pf = load_fragA(pw, 32, 0, lLow, lHi);
    o0 = WMMA_BF16(pf, load_fragB(sVt, 32, 0, 0, lLow, lHi), o0);
    o1 = WMMA_BF16(pf, load_fragB(sVt, 32, 16, 0, lLow, lHi), o1);
    o2 = WMMA_BF16(pf, load_fragB(sVt, 32, 32, 0, lLow, lHi), o2);
    o3 = WMMA_BF16(pf, load_fragB(sVt, 32, 48, 0, lLow, lHi), o3);
  }
#pragma unroll
  for (int r = 0; r < 8; ++r) {
    float inv = 1.f / lrun[r];
    size_t row = rowBase + q0 + lHi * 8 + r;
    __bf16* cp = ctx + row * kD + h * kHD + lLow;
    cp[0]  = (__bf16)(o0[r] * inv);
    cp[16] = (__bf16)(o1[r] * inv);
    cp[32] = (__bf16)(o2[r] * inv);
    cp[48] = (__bf16)(o3[r] * inv);
  }
}

// ---------------- gating: logits, softmax, top-2 ----------------
__global__ __launch_bounds__(256) void gate_kernel(const __bf16* __restrict__ h2,
                                                   const float* __restrict__ gw,
                                                   const float* __restrict__ gb,
                                                   float* __restrict__ ew) {
  const int wave = threadIdx.x >> 5, lane = threadIdx.x & 31;
  const int t = blockIdx.x * 8 + wave;
  float acc[kE] = {};
  for (int d = lane; d < kD; d += 32) {
    float xv = (float)h2[(size_t)t * kD + d];
#pragma unroll
    for (int e = 0; e < kE; ++e) acc[e] += xv * gw[e * kD + d];
  }
#pragma unroll
  for (int e = 0; e < kE; ++e)
#pragma unroll
    for (int m = 1; m < 32; m <<= 1) acc[e] += bcast_xor_f(acc[e], m);
  if (lane == 0) {
    float p[kE];
    float mx = -3e38f;
    for (int e = 0; e < kE; ++e) { p[e] = acc[e] + gb[e]; mx = fmaxf(mx, p[e]); }
    float sum = 0.f;
    for (int e = 0; e < kE; ++e) { p[e] = __expf(p[e] - mx); sum += p[e]; }
    float inv = 1.f / sum;
    for (int e = 0; e < kE; ++e) p[e] *= inv;
    int i1 = 0;
    for (int e = 1; e < kE; ++e) if (p[e] > p[i1]) i1 = e;
    int i2 = (i1 == 0) ? 1 : 0;
    for (int e = 0; e < kE; ++e) if (e != i1 && p[e] > p[i2]) i2 = e;
    float tw = fmaxf(p[i1] + p[i2], 1e-9f);
    for (int e = 0; e < kE; ++e) ew[(size_t)t * kE + e] = 0.f;
    ew[(size_t)t * kE + i1] = p[i1] / tw;
    ew[(size_t)t * kE + i2] = p[i2] / tw;
  }
}

// ---------------- expert compaction (wave e scans tokens, prefix-scan, pad to 128) ----------------
__global__ __launch_bounds__(256) void compact_kernel(const float* __restrict__ ew,
                                                      int* __restrict__ tokList,
                                                      float* __restrict__ tokW,
                                                      int* __restrict__ meta) {
  const int e = threadIdx.x >> 5, lane = threadIdx.x & 31;
  int base = 0;
  for (int t0 = 0; t0 < kT; t0 += 32) {
    int t = t0 + lane;
    float w = ew[(size_t)t * kE + e];
    int flag = (w > 0.f) ? 1 : 0;
    int v = flag;  // inclusive scan across 32 lanes
#pragma unroll
    for (int off = 1; off < 32; off <<= 1) {
      int src = lane - off;
      int vv = bperm_i(src < 0 ? lane : src, v);
      if (lane >= off) v += vv;
    }
    int total = bperm_i(31, v);
    if (flag) {
      int pos = base + v - 1;
      tokList[e * kT + pos] = t;
      tokW[e * kT + pos] = w;
    }
    base += total;
  }
  int padTo = (base + 127) & ~127;
  for (int p = base + lane; p < padTo; p += 32) { tokList[e * kT + p] = 0; tokW[e * kT + p] = 0.f; }
  if (lane == 0) meta[e] = padTo;
  __syncthreads();
  if (threadIdx.x == 0) {
    int o = 0;
    for (int i = 0; i < kE; ++i) { meta[8 + i] = o; o += meta[i]; }
    meta[16] = o;  // offsets[8] = total padded slots
  }
}

// ---------------- MoE expert GEMM 1: hid = silu(gather(h2) @ w1[e]^T + b1[e]) ----------------
__global__ __launch_bounds__(256) void moe_w1_kernel(const __bf16* __restrict__ h2,
                                                     const __bf16* __restrict__ w1bf,
                                                     const float* __restrict__ b1,
                                                     const int* __restrict__ tokList,
                                                     const int* __restrict__ meta,
                                                     __bf16* __restrict__ hid) {
  const int* offsets = meta + 8;
  const int row0 = blockIdx.y * 128;
  if (row0 >= offsets[kE]) return;
  int e = 0;
  while (e < kE - 1 && row0 >= offsets[e + 1]) ++e;
  __shared__ int sTok[128];
  __shared__ __bf16 sA[128 * 32];
  __shared__ __bf16 sB[128 * 32];
  const int tid = threadIdx.x;
  if (tid < 128) sTok[tid] = tokList[e * kT + (row0 - offsets[e]) + tid];
  const int wave = tid >> 5, lane = tid & 31, lLow = lane & 15, lHi = lane >> 4;
  const int nBlk = blockIdx.x * 128;
  const int mW = (wave >> 1) * 32, nW = (wave & 1) * 64;
  const __bf16* W = w1bf + (size_t)e * kFF * kD;
  v8f acc[2][4] = {};
  for (int k0 = 0; k0 < kD; k0 += 32) {
    __syncthreads();
    stage_tile_gather(sA, h2, sTok, kD, k0, tid);
    stage_tile(sB, W, nBlk, kD, k0, tid);
    if (k0 + 32 < kD) {
      int idx = tid * 8, r = idx >> 5, c = idx & 31;
      __builtin_prefetch(&W[(size_t)(nBlk + r) * kD + k0 + 32 + c], 0, 0);
    }
    async_wait0();
    __syncthreads();
    wave_mma(sA, sB, mW, nW, lLow, lHi, acc);
  }
  const int colBase = nBlk + nW;
#pragma unroll
  for (int mi = 0; mi < 2; ++mi)
#pragma unroll
    for (int j = 0; j < 4; ++j) {
      int col = colBase + j * 16 + lLow;
      float bv = b1[e * kFF + col];
#pragma unroll
      for (int r = 0; r < 8; ++r) {
        int slot = row0 + mW + mi * 16 + lHi * 8 + r;
        float v = acc[mi][j][r] + bv;
        float h = v / (1.f + __expf(-v));  // silu
        hid[(size_t)slot * kFF + col] = (__bf16)h;
      }
    }
}

// ---------------- MoE expert GEMM 2: out[token] += w * (hid @ w2[e]^T + b2[e]) ----------------
__global__ __launch_bounds__(256) void moe_w2_kernel(const __bf16* __restrict__ hid,
                                                     const __bf16* __restrict__ w2bf,
                                                     const float* __restrict__ b2,
                                                     const int* __restrict__ tokList,
                                                     const float* __restrict__ tokW,
                                                     const int* __restrict__ meta,
                                                     float* __restrict__ out) {
  const int* offsets = meta + 8;
  const int row0 = blockIdx.y * 128;
  if (row0 >= offsets[kE]) return;
  int e = 0;
  while (e < kE - 1 && row0 >= offsets[e + 1]) ++e;
  __shared__ __bf16 sA[128 * 32];
  __shared__ __bf16 sB[128 * 32];
  const int tid = threadIdx.x, wave = tid >> 5, lane = tid & 31;
  const int lLow = lane & 15, lHi = lane >> 4;
  const int nBlk = blockIdx.x * 128;
  const int mW = (wave >> 1) * 32, nW = (wave & 1) * 64;
  const __bf16* W = w2bf + (size_t)e * kD * kFF;
  v8f acc[2][4] = {};
  for (int k0 = 0; k0 < kFF; k0 += 32) {
    __syncthreads();
    stage_tile(sA, hid, row0, kFF, k0, tid);
    stage_tile(sB, W, nBlk, kFF, k0, tid);
    if (k0 + 32 < kFF) {
      int idx = tid * 8, r = idx >> 5, c = idx & 31;
      __builtin_prefetch(&W[(size_t)(nBlk + r) * kFF + k0 + 32 + c], 0, 0);
    }
    async_wait0();
    __syncthreads();
    wave_mma(sA, sB, mW, nW, lLow, lHi, acc);
  }
  const int colBase = nBlk + nW;
#pragma unroll
  for (int mi = 0; mi < 2; ++mi)
#pragma unroll
    for (int r = 0; r < 8; ++r) {
      int slot = row0 + mW + mi * 16 + lHi * 8 + r;
      int li = e * kT + (slot - offsets[e]);
      int token = tokList[li];
      float w = tokW[li];
      if (w != 0.f) {
#pragma unroll
        for (int j = 0; j < 4; ++j) {
          int col = colBase + j * 16 + lLow;
          unsafeAtomicAdd(&out[(size_t)token * kD + col], w * (acc[mi][j][r] + b2[e * kD + col]));
        }
      }
    }
}

// ---------------- host-side launch ----------------
extern "C" void kernel_launch(void* const* d_in, const int* in_sizes, int n_in,
                              void* d_out, int out_size, void* d_ws, size_t ws_size,
                              hipStream_t stream) {
  const float* x         = (const float*)d_in[0];
  const float* ln1_g     = (const float*)d_in[1];
  const float* ln1_b     = (const float*)d_in[2];
  const float* in_proj_w = (const float*)d_in[3];
  const float* in_proj_b = (const float*)d_in[4];
  const float* out_w     = (const float*)d_in[5];
  const float* out_b     = (const float*)d_in[6];
  const float* ln2_g     = (const float*)d_in[7];
  const float* ln2_b     = (const float*)d_in[8];
  const float* gate_w    = (const float*)d_in[9];
  const float* gate_b    = (const float*)d_in[10];
  const float* w1        = (const float*)d_in[11];
  const float* b1        = (const float*)d_in[12];
  const float* w2        = (const float*)d_in[13];
  const float* b2        = (const float*)d_in[14];
  float* out = (float*)d_out;
  char* ws = (char*)d_ws;

  __bf16* inprojB = (__bf16*)(ws + OFF_INPROJ);
  __bf16* outwB   = (__bf16*)(ws + OFF_OUTW);
  __bf16* w1B     = (__bf16*)(ws + OFF_W1);
  __bf16* w2B     = (__bf16*)(ws + OFF_W2);
  __bf16* h1B     = (__bf16*)(ws + OFF_H1);
  __bf16* qkvB    = (__bf16*)(ws + OFF_QKVB);
  __bf16* ctxB    = (__bf16*)(ws + OFF_CTX);
  __bf16* h2B     = (__bf16*)(ws + OFF_H2);
  float*  ewP     = (float*)(ws + OFF_EW);
  int*    tokL    = (int*)(ws + OFF_TOKL);
  float*  tokWp   = (float*)(ws + OFF_TOKW);
  int*    meta    = (int*)(ws + OFF_META);
  __bf16* hidB    = (__bf16*)(ws + OFF_HID);

  // 1) cast weights to bf16 once per launch
  f2bf_kernel<<<(kQKV * kD) / 1024, 256, 0, stream>>>(in_proj_w, inprojB, kQKV * kD);
  f2bf_kernel<<<(kD * kD) / 1024, 256, 0, stream>>>(out_w, outwB, kD * kD);
  f2bf_kernel<<<(kE * kFF * kD) / 1024, 256, 0, stream>>>(w1, w1B, kE * kFF * kD);
  f2bf_kernel<<<(kE * kD * kFF) / 1024, 256, 0, stream>>>(w2, w2B, kE * kD * kFF);
  // 2) LN1
  ln_kernel<<<kT, 256, 0, stream>>>(x, ln1_g, ln1_b, h1B);
  // 3) QKV projection
  gemm_kernel<true><<<dim3(kQKV / 128, kT / 128), 256, 0, stream>>>(
      h1B, inprojB, in_proj_b, nullptr, qkvB, nullptr, kD, kQKV);
  // 4) attention
  attn_kernel<<<dim3(kSeq / 128, kBatch * kH), 256, 0, stream>>>(qkvB, ctxB);
  // 5) out projection + residual -> d_out
  gemm_kernel<false><<<dim3(kD / 128, kT / 128), 256, 0, stream>>>(
      ctxB, outwB, out_b, x, nullptr, out, kD, kD);
  // 6) LN2
  ln_kernel<<<kT, 256, 0, stream>>>(out, ln2_g, ln2_b, h2B);
  // 7) gating + 8) compaction
  gate_kernel<<<kT / 8, 256, 0, stream>>>(h2B, gate_w, gate_b, ewP);
  compact_kernel<<<1, 256, 0, stream>>>(ewP, tokL, tokWp, meta);
  // 9) expert FFN
  moe_w1_kernel<<<dim3(kFF / 128, kSlotCap / 128), 256, 0, stream>>>(
      h2B, w1B, b1, tokL, meta, hidB);
  moe_w2_kernel<<<dim3(kD / 128, kSlotCap / 128), 256, 0, stream>>>(
      hidB, w2B, b2, tokL, tokWp, meta, out);

  (void)in_sizes; (void)n_in; (void)out_size; (void)ws_size;
}
